// AttentionHead_11871289606295
// MI455X (gfx1250) — matrix operand order repaired and checked by reading, not compiled, not run on hardware
//
#include <hip/hip_runtime.h>
#include <hip/hip_bf16.h>
#include <math.h>

// ---------------------------------------------------------------------------
// Types for CDNA5 WMMA (wave32)
// ---------------------------------------------------------------------------
typedef __bf16 bf16_t;
typedef bf16_t v8bf  __attribute__((ext_vector_type(8)));
typedef bf16_t v16bf __attribute__((ext_vector_type(16)));
typedef float  v8f   __attribute__((ext_vector_type(8)));

#define D_MODEL 768
#define N3D     2304
#define N_HEADS 12
#define BATCH   2
#define SEQ     2048
#define DH      64
#define NTOK    (BATCH * SEQ)     // 4096
#define BHN     (BATCH * N_HEADS) // 24
#define NQT     (SEQ / 16)        // 128 q-tiles per head
#define ATT_SCALE 0.03608439182435161f  // 1/sqrt(768)

__device__ __forceinline__ bf16_t f2bf(float f) {
  unsigned u = __builtin_bit_cast(unsigned, f);
  unsigned r = (u >> 16) & 1u;
  u += 0x7fffu + r;                       // round-to-nearest-even
  unsigned short h = (unsigned short)(u >> 16);
  return __builtin_bit_cast(bf16_t, h);
}

__device__ __forceinline__ v8f wmma_bf16(v16bf a, v16bf b, v8f c) {
  // D = A(16x32 bf16) * B(32x16 bf16) + C(16x16 f32)
  return __builtin_amdgcn_wmma_f32_16x16x32_bf16(
      false, a, false, b, (short)0, c, false, false);
}

// A fragment: row-major A[m][k], ld elements.
// lane<16: row m=lane,     K = {k0..k0+7, k0+16..k0+23}
// lane>=16: row m=lane-16, K = {k0+8..k0+15, k0+24..k0+31}
__device__ __forceinline__ v16bf load_frag_a(const bf16_t* A, int ld,
                                             int row0, int k0, int lane) {
  int m  = lane & 15;
  int kh = (lane >> 4) * 8;
  const bf16_t* p = A + (size_t)(row0 + m) * ld + k0 + kh;
  v8bf lo = *(const v8bf*)(p);
  v8bf hi = *(const v8bf*)(p + 16);
  return __builtin_shufflevector(lo, hi, 0,1,2,3,4,5,6,7,8,9,10,11,12,13,14,15);
}

// B fragment from Bt stored [n][k] (K contiguous), ld elements.
// lane<16: col n=lane, K={k0..k0+7, k0+16..k0+23} ; lane>=16: col n=lane-16, K={k0+8..k0+15, k0+24..k0+31}
__device__ __forceinline__ v16bf load_frag_b(const bf16_t* Bt, int ld,
                                             int col0, int k0, int lane) {
  int n  = lane & 15;
  int kb = k0 + (lane >> 4) * 8;
  const bf16_t* p = Bt + (size_t)(col0 + n) * ld + kb;
  v8bf lo = *(const v8bf*)(p);
  v8bf hi = *(const v8bf*)(p + 16);
  return __builtin_shufflevector(lo, hi, 0,1,2,3,4,5,6,7,8,9,10,11,12,13,14,15);
}

// C/D fragment element (r, lane): m = r + 8*(lane>>4), n = lane&15

// ---------------------------------------------------------------------------
// Conversion kernels
// ---------------------------------------------------------------------------
__global__ void k_cvt_bf16(const float* __restrict__ in, bf16_t* __restrict__ out, int n) {
  int i = blockIdx.x * blockDim.x + threadIdx.x;
  if (i < n) out[i] = f2bf(in[i]);
}

// in[k][n] (KxN fp32) -> out[n][k] (NxK bf16)
__global__ void k_cvt_transpose(const float* __restrict__ in, bf16_t* __restrict__ out,
                                int K, int N) {
  int o = blockIdx.x * blockDim.x + threadIdx.x;
  if (o < K * N) {
    int n = o / K;
    int k = o - n * K;
    out[o] = f2bf(in[(size_t)k * N + n]);
  }
}

// ---------------------------------------------------------------------------
// GEMM1: qkv = x @ W_in + b_in, scatter into Q / K / Vt (bf16)
// A: xb [4096][768], Bt: WtIn [2304][768]
// block: 256 thr (8 waves), wave tile 32x64 (2x4), block tile 64x256
// ---------------------------------------------------------------------------
__global__ void k_gemm_qkv(const bf16_t* __restrict__ A, const bf16_t* __restrict__ Bt,
                           const float* __restrict__ bias,
                           bf16_t* __restrict__ Qb, bf16_t* __restrict__ Kb,
                           bf16_t* __restrict__ Vt) {
  const int tid  = threadIdx.x;
  const int lane = tid & 31;
  const int wid  = tid >> 5;
  const int row0 = blockIdx.x * 64  + (wid & 1) * 32;
  const int col0 = blockIdx.y * 256 + (wid >> 1) * 64;

  v8f acc[2][4] = {};
  for (int kb = 0; kb < D_MODEL; kb += 32) {
    v16bf a0 = load_frag_a(A, D_MODEL, row0,      kb, lane);
    v16bf a1 = load_frag_a(A, D_MODEL, row0 + 16, kb, lane);
    v16bf bf[4];
#pragma unroll
    for (int j = 0; j < 4; ++j)
      bf[j] = load_frag_b(Bt, D_MODEL, col0 + j * 16, kb, lane);
#pragma unroll
    for (int j = 0; j < 4; ++j) {
      acc[0][j] = wmma_bf16(a0, bf[j], acc[0][j]);
      acc[1][j] = wmma_bf16(a1, bf[j], acc[1][j]);
    }
  }

  const int mofs = (lane >> 4) * 8;
  const int n    = lane & 15;
#pragma unroll
  for (int i = 0; i < 2; ++i)
#pragma unroll
    for (int j = 0; j < 4; ++j)
#pragma unroll
      for (int r = 0; r < 8; ++r) {
        int t  = row0 + i * 16 + r + mofs;
        int cc = col0 + j * 16 + n;
        float val = acc[i][j][r] + bias[cc];
        int which = cc / D_MODEL;
        int dm    = cc - which * D_MODEL;
        int h  = dm >> 6;
        int dh = dm & 63;
        int bb = t >> 11;          // / SEQ
        int ss = t & (SEQ - 1);
        size_t hb = (size_t)(bb * N_HEADS + h);
        bf16_t v = f2bf(val);
        if (which == 0)      Qb[(hb * SEQ + ss) * DH + dh] = v;
        else if (which == 1) Kb[(hb * SEQ + ss) * DH + dh] = v;
        else                 Vt[(hb * DH + dh) * SEQ + ss] = v;
      }
}

__device__ __forceinline__ void smerge(float& m, float& l, float m2, float l2) {
  float mn = fmaxf(m, m2);
  if (mn == -__builtin_inff()) return;
  float ea = (m  == -__builtin_inff()) ? 0.f : __expf(m  - mn);
  float eb = (m2 == -__builtin_inff()) ? 0.f : __expf(m2 - mn);
  l = l * ea + l2 * eb;
  m = mn;
}

// ---------------------------------------------------------------------------
// Pass 1: per-key-column softmax stats over the query axis (axis=-2 softmax).
// grid = BHN * NQT blocks; block handles one (bh, key-tile of 16 columns).
// 8 waves split the q-tiles (q >= key due to causal mask).
// ---------------------------------------------------------------------------
__global__ void k_col_stats(const bf16_t* __restrict__ Qb, const bf16_t* __restrict__ Kb,
                            float* __restrict__ mstat, float* __restrict__ lstat) {
  const int bh = blockIdx.x / NQT;
  const int kt = blockIdx.x - bh * NQT;
  const int n0 = kt * 16;
  const int tid  = threadIdx.x;
  const int lane = tid & 31;
  const int wid  = tid >> 5;
  const int mofs = (lane >> 4) * 8;
  const int kg   = n0 + (lane & 15);

  const bf16_t* Qh = Qb + (size_t)bh * SEQ * DH;
  const bf16_t* Kh = Kb + (size_t)bh * SEQ * DH;

  // key-tile fragments are invariant over the q loop: hoist
  const v16bf kf0 = load_frag_b(Kh, DH, n0, 0,  lane);
  const v16bf kf1 = load_frag_b(Kh, DH, n0, 32, lane);

  const float NEG = -__builtin_inff();
  float m_run = NEG, l_run = 0.f;

  for (int qt = kt + wid; qt < NQT; qt += 8) {
    int q0 = qt * 16;
    v8f s = {};
    s = wmma_bf16(load_frag_a(Qh, DH, q0, 0,  lane), kf0, s);
    s = wmma_bf16(load_frag_a(Qh, DH, q0, 32, lane), kf1, s);

    float sv[8];
    float tmax = NEG;
#pragma unroll
    for (int r = 0; r < 8; ++r) {
      int qg = q0 + r + mofs;
      float x = (qg >= kg) ? s[r] * ATT_SCALE : NEG;
      sv[r] = x;
      tmax = fmaxf(tmax, x);
    }
    if (tmax != NEG) {
      float mn = fmaxf(m_run, tmax);
      float so = (m_run == NEG) ? 0.f : __expf(m_run - mn);
      float sum = 0.f;
#pragma unroll
      for (int r = 0; r < 8; ++r)
        if (sv[r] != NEG) sum += __expf(sv[r] - mn);
      l_run = l_run * so + sum;
      m_run = mn;
    }
  }

  // merge lane pairs (rows 0-7 with rows 8-15 of each column)
  float m2 = __shfl_xor(m_run, 16, 32);
  float l2 = __shfl_xor(l_run, 16, 32);
  smerge(m_run, l_run, m2, l2);

  __shared__ float sm[8][16];
  __shared__ float sl[8][16];
  if (lane < 16) { sm[wid][lane] = m_run; sl[wid][lane] = l_run; }
  __syncthreads();
  if (tid < 16) {
    float m = NEG, l = 0.f;
#pragma unroll
    for (int w = 0; w < 8; ++w) smerge(m, l, sm[w][tid], sl[w][tid]);
    mstat[(size_t)bh * SEQ + n0 + tid] = m;
    lstat[(size_t)bh * SEQ + n0 + tid] = l;
  }
}

// ---------------------------------------------------------------------------
// Pass 2: out = normalize(P) @ V per head. One wave per q-tile (16 rows).
// P tile transposed C-frag -> A-frag via a private 16x32 bf16 LDS buffer.
// grid = BHN * (NQT/8); 8 waves per block, independent.
// ---------------------------------------------------------------------------
__global__ void k_attn_pv(const bf16_t* __restrict__ Qb, const bf16_t* __restrict__ Kb,
                          const bf16_t* __restrict__ Vt,
                          const float* __restrict__ mstat, const float* __restrict__ lstat,
                          bf16_t* __restrict__ attn_out) {
  const int bh  = blockIdx.x / (NQT / 8);
  const int grp = blockIdx.x - bh * (NQT / 8);
  const int tid  = threadIdx.x;
  const int lane = tid & 31;
  const int wid  = tid >> 5;
  const int qt   = grp * 8 + wid;
  const int q0   = qt * 16;
  const int mofs = (lane >> 4) * 8;
  const int nn   = lane & 15;

  const bf16_t* Qh = Qb + (size_t)bh * SEQ * DH;
  const bf16_t* Kh = Kb + (size_t)bh * SEQ * DH;
  const bf16_t* Vh = Vt + (size_t)bh * DH * SEQ;
  const float*  mr = mstat + (size_t)bh * SEQ;
  const float*  lr = lstat + (size_t)bh * SEQ;

  __shared__ __align__(16) bf16_t Pt[8][16 * 32];
  bf16_t* myP = &Pt[wid][0];

  // q-tile fragments are invariant over the key-chunk loop: hoist
  const v16bf qf0 = load_frag_a(Qh, DH, q0, 0,  lane);
  const v16bf qf1 = load_frag_a(Qh, DH, q0, 32, lane);

  v8f o[4] = {};
  const int nchunks = qt / 2 + 1;   // 32-key chunks covering k <= q0+15
  for (int c = 0; c < nchunks; ++c) {
    int kb = c * 32;
#pragma unroll
    for (int half = 0; half < 2; ++half) {
      int n0 = kb + half * 16;
      v8f s = {};
      s = wmma_bf16(qf0, load_frag_b(Kh, DH, n0, 0,  lane), s);
      s = wmma_bf16(qf1, load_frag_b(Kh, DH, n0, 32, lane), s);
      int kg = n0 + nn;
      float mk = mr[kg];
      float lk = lr[kg];
      float li = (lk > 0.f) ? 1.0f / lk : 0.f;
#pragma unroll
      for (int r = 0; r < 8; ++r) {
        int qg = q0 + r + mofs;
        float p = (qg >= kg) ? __expf(s[r] * ATT_SCALE - mk) * li : 0.f;
        myP[(r + mofs) * 32 + half * 16 + nn] = f2bf(p);
      }
    }
    // reload P as an A fragment (16x32, ld=32) and do 4 PV WMMAs (Dh=64)
    v16bf pa = load_frag_a(myP, 32, 0, 0, lane);
#pragma unroll
    for (int d = 0; d < 4; ++d)
      o[d] = wmma_bf16(pa, load_frag_b(Vh, SEQ, d * 16, kb, lane), o[d]);
  }

  const int b = bh / N_HEADS;
  const int h = bh - b * N_HEADS;
#pragma unroll
  for (int d = 0; d < 4; ++d)
#pragma unroll
    for (int r = 0; r < 8; ++r) {
      int t  = b * SEQ + q0 + r + mofs;
      int dm = h * DH + d * 16 + nn;
      attn_out[(size_t)t * D_MODEL + dm] = f2bf(o[d][r]);
    }
}

// ---------------------------------------------------------------------------
// GEMM2: out = attn_out @ W_out + b_out  (fp32 result)
// wave tile 32x64 (2x4), block tile 64x256
// ---------------------------------------------------------------------------
__global__ void k_gemm_out(const bf16_t* __restrict__ A, const bf16_t* __restrict__ Bt,
                           const float* __restrict__ bias, float* __restrict__ out) {
  const int tid  = threadIdx.x;
  const int lane = tid & 31;
  const int wid  = tid >> 5;
  const int row0 = blockIdx.x * 64  + (wid & 1) * 32;
  const int col0 = blockIdx.y * 256 + (wid >> 1) * 64;

  v8f acc[2][4] = {};
  for (int kb = 0; kb < D_MODEL; kb += 32) {
    v16bf a0 = load_frag_a(A, D_MODEL, row0,      kb, lane);
    v16bf a1 = load_frag_a(A, D_MODEL, row0 + 16, kb, lane);
    v16bf bf[4];
#pragma unroll
    for (int j = 0; j < 4; ++j)
      bf[j] = load_frag_b(Bt, D_MODEL, col0 + j * 16, kb, lane);
#pragma unroll
    for (int j = 0; j < 4; ++j) {
      acc[0][j] = wmma_bf16(a0, bf[j], acc[0][j]);
      acc[1][j] = wmma_bf16(a1, bf[j], acc[1][j]);
    }
  }

  const int mofs = (lane >> 4) * 8;
  const int n    = lane & 15;
#pragma unroll
  for (int i = 0; i < 2; ++i)
#pragma unroll
    for (int j = 0; j < 4; ++j)
#pragma unroll
      for (int r = 0; r < 8; ++r) {
        int t  = row0 + i * 16 + r + mofs;
        int cc = col0 + j * 16 + n;
        out[(size_t)t * D_MODEL + cc] = acc[i][j][r] + bias[cc];
      }
}

// ---------------------------------------------------------------------------
// Host launcher
// ---------------------------------------------------------------------------
static inline size_t align256(size_t x) { return (x + 255) & ~(size_t)255; }

extern "C" void kernel_launch(void* const* d_in, const int* in_sizes, int n_in,
                              void* d_out, int out_size, void* d_ws, size_t ws_size,
                              hipStream_t stream) {
  (void)in_sizes; (void)n_in; (void)out_size; (void)ws_size;

  const float* x     = (const float*)d_in[0];
  // d_in[1] is the tril mask (int32) -- causality is applied analytically.
  const float* W_in  = (const float*)d_in[2];
  const float* b_in  = (const float*)d_in[3];
  const float* W_out = (const float*)d_in[4];
  const float* b_out = (const float*)d_in[5];
  float* out = (float*)d_out;

  char* w = (char*)d_ws;
  bf16_t* xb    = (bf16_t*)w; w += align256((size_t)NTOK * D_MODEL * 2);
  bf16_t* WtIn  = (bf16_t*)w; w += align256((size_t)N3D * D_MODEL * 2);
  bf16_t* WtOut = (bf16_t*)w; w += align256((size_t)D_MODEL * D_MODEL * 2);
  bf16_t* Qb    = (bf16_t*)w; w += align256((size_t)BHN * SEQ * DH * 2);
  bf16_t* Kb    = (bf16_t*)w; w += align256((size_t)BHN * SEQ * DH * 2);
  bf16_t* Vt    = (bf16_t*)w; w += align256((size_t)BHN * DH * SEQ * 2);
  float*  mst   = (float*)w;  w += align256((size_t)BHN * SEQ * 4);
  float*  lst   = (float*)w;  w += align256((size_t)BHN * SEQ * 4);
  bf16_t* attn  = (bf16_t*)w; w += align256((size_t)NTOK * D_MODEL * 2);

  // 1) precision conversion / weight transposes
  k_cvt_bf16<<<(NTOK * D_MODEL) / 256, 256, 0, stream>>>(x, xb, NTOK * D_MODEL);
  k_cvt_transpose<<<(D_MODEL * N3D) / 256, 256, 0, stream>>>(W_in, WtIn, D_MODEL, N3D);
  k_cvt_transpose<<<(D_MODEL * D_MODEL) / 256, 256, 0, stream>>>(W_out, WtOut, D_MODEL, D_MODEL);

  // 2) qkv projection (+ bias) with head scatter
  k_gemm_qkv<<<dim3(NTOK / 64, N3D / 256), 256, 0, stream>>>(xb, WtIn, b_in, Qb, Kb, Vt);

  // 3) column-softmax stats (softmax over query axis)
  k_col_stats<<<BHN * NQT, 256, 0, stream>>>(Qb, Kb, mst, lst);

  // 4) normalized P @ V
  k_attn_pv<<<BHN * (NQT / 8), 256, 0, stream>>>(Qb, Kb, Vt, mst, lst, attn);

  // 5) output projection (+ bias) -> fp32
  k_gemm_out<<<dim3(NTOK / 64, D_MODEL / 256), 256, 0, stream>>>(attn, WtOut, b_out, out);
}
